// TGN_80573586473148
// MI455X (gfx1250) — compile-verified, matrix operand order checked by reading, not run on hardware
//
#include <hip/hip_runtime.h>
#include <hip/hip_bf16.h>

// ---------------------------------------------------------------------------
// S=16384 timesteps, D=1024 input, H=1024 hidden LSTM + scalar recurrences.
//   Phase 0: one-shot fp32 -> bf16 conversion of inputs / W_ih (memory bound)
//   Phase 1: xg[S,4H] = inputs @ W_ih^T + (b_ih+b_hh)   -> WMMA bf16 GEMM,
//            global->LDS staging via TDM tensor_load_to_lds (async DMA)
//   Phase 2: sequential LSTM scan (16 WGs, grid-synced), W_hh L2-resident
//   Phase 3: alpha = sigmoid(h . W_fc + b_fc)
//   Phase 4: affine-recurrence parallel scans for (a_seq, y_seq) + loss
// Output layout: [loss(1), y_seq(16383), a_seq(16383), u(16383)]
// ---------------------------------------------------------------------------

#define S_LEN   16384
#define DIM     1024
#define HID     1024
#define G4      4096          // 4*HID
#define NWG     16            // workgroups in the scan kernel
#define REC_N   16383         // S_LEN - 1

typedef __attribute__((ext_vector_type(16))) __bf16    v16bf;
typedef __attribute__((ext_vector_type(8)))  float     v8f;
typedef __attribute__((ext_vector_type(4)))  unsigned  tdm_g0;
typedef __attribute__((ext_vector_type(8)))  int       tdm_g1;
typedef __attribute__((ext_vector_type(4)))  int       tdm_g2;

union FragU { unsigned u[8]; v16bf v; };

static __device__ __forceinline__ unsigned short to_bf16_bits(float f) {
  union { __bf16 h; unsigned short u; } c;
  c.h = (__bf16)f;                       // hardware cvt path
  return c.u;
}

// Load one 16x32 bf16 A/B fragment from an LDS tile (row-major, ld=32 bf16).
// Layout per CDNA5 ISA 7.12.2 (16-bit A-matrix 16x32):
//   lanes 0-15: M=lane, VGPR p<4 -> K=2p ; p>=4 -> K=16+2(p-4)
//   lanes 16-31: same rows, K offset +8 in each half.
static __device__ __forceinline__ v16bf load_frag(const unsigned short* buf0,
                                                  int row, int lane) {
  FragU f;
  const int half = (lane >> 4) & 1;
  const unsigned short* r = buf0 + row * 32;
#pragma unroll
  for (int p = 0; p < 8; ++p) {
    int k = (p < 4) ? (8 * half + 2 * p) : (16 + 8 * half + 2 * (p - 4));
    f.u[p] = *(const unsigned*)(r + k);
  }
  return f.v;
}

// --------------------------- TDM descriptor build --------------------------
// D# group 0 (ISA 8.3): count=1 | lds_addr[63:32] | global_addr[120:64] |
//                       type=2 in [127:126]
static __device__ __forceinline__ tdm_g0 make_g0(unsigned lds_addr,
                                                 unsigned long long gaddr) {
  tdm_g0 g;
  g[0] = 1u;                                            // count=1, user load
  g[1] = lds_addr;
  g[2] = (unsigned)(gaddr & 0xffffffffull);
  g[3] = (unsigned)((gaddr >> 32) & 0x01ffffffull) | (2u << 30);
  return g;
}
// D# group 1 (ISA 8.4): 2D tile, data_size=1 (2B), no padding/iterate.
static __device__ __forceinline__ tdm_g1 make_g1_2d(unsigned tensor_d0,
                                                    unsigned tensor_d1,
                                                    unsigned tile_d0,
                                                    unsigned tile_d1,
                                                    unsigned long long stride0) {
  tdm_g1 g;
  g[0] = (int)(1u << 16);                               // data_size = 2 bytes
  g[1] = (int)((tensor_d0 & 0xffffu) << 16);            // dim0[15:0]
  g[2] = (int)((tensor_d0 >> 16) | ((tensor_d1 & 0xffffu) << 16));
  g[3] = (int)((tensor_d1 >> 16) | ((tile_d0 & 0xffffu) << 16));
  g[4] = (int)(tile_d1 & 0xffffu);                      // tile_dim2 = 0
  g[5] = (int)(unsigned)(stride0 & 0xffffffffull);
  g[6] = (int)(unsigned)((stride0 >> 32) & 0xffffull);  // dim1_stride = 0
  g[7] = 0;
  return g;
}

#if defined(__gfx1250__) && __has_builtin(__builtin_amdgcn_tensor_load_to_lds)
#define USE_TDM 1
#else
#define USE_TDM 0
#endif

// ---------------------------------------------------------------------------
// Kernel 0: init sync words (+ h double buffer) each launch (determinism)
// ---------------------------------------------------------------------------
__global__ void init_kernel(int* syncs, float* hglob) {
  int t = threadIdx.x;
  if (t < 2) syncs[t] = 0;
  for (int i = t; i < 2 * HID; i += blockDim.x) hglob[i] = 0.f;
}

// ---------------------------------------------------------------------------
// Kernel 0b: one-shot fp32 -> bf16 conversion (n multiple of 8)
// ---------------------------------------------------------------------------
__global__ void __launch_bounds__(256)
cvt_bf16(const float* __restrict__ src, unsigned short* __restrict__ dst, int n) {
  int i = (blockIdx.x * 256 + threadIdx.x) * 8;
  if (i + 8 <= n) {
    float4 a = *(const float4*)(src + i);
    float4 b = *(const float4*)(src + i + 4);
    uint4 o;
    o.x = (unsigned)to_bf16_bits(a.x) | ((unsigned)to_bf16_bits(a.y) << 16);
    o.y = (unsigned)to_bf16_bits(a.z) | ((unsigned)to_bf16_bits(a.w) << 16);
    o.z = (unsigned)to_bf16_bits(b.x) | ((unsigned)to_bf16_bits(b.y) << 16);
    o.w = (unsigned)to_bf16_bits(b.z) | ((unsigned)to_bf16_bits(b.w) << 16);
    *(uint4*)(dst + i) = o;
  }
}

// ---------------------------------------------------------------------------
// Kernel 1: xg = inputs(16384x1024) @ W_ih^T(1024x4096) + (b_ih+b_hh)
// 128x128 tile per WG (256 threads, 8 waves; wave tile 64x32 = 4x2 WMMAs).
// Operands pre-converted to bf16; staged to LDS by the Tensor Data Mover
// (one 2D D# per tile, double buffered, s_wait_tensorcnt + barrier).
// ---------------------------------------------------------------------------
__global__ void __launch_bounds__(256)
gemm_xg(const unsigned short* __restrict__ Abf,  // inputs bf16 [16384,1024]
        const unsigned short* __restrict__ Wbf,  // W_ih  bf16 [4096,1024]
        const float* __restrict__ b_ih,
        const float* __restrict__ b_hh,
        float* __restrict__ xg)                  // [16384,4096]
{
  __shared__ __align__(16) unsigned short sA[2][128 * 32];
  __shared__ __align__(16) unsigned short sB[2][128 * 32];

  const int tid  = threadIdx.x;
  const int lane = tid & 31;
  const int wave = tid >> 5;            // 0..7
  const int wm   = wave >> 2;           // 0..1  -> 64-row strip
  const int wn   = wave & 3;            // 0..3  -> 32-col strip
  const int n0   = blockIdx.x * 128;    // N tile (4096/128 = 32)
  const int m0   = blockIdx.y * 128;    // M tile (16384/128 = 128)

  v8f acc[4][2];
#pragma unroll
  for (int mt = 0; mt < 4; ++mt)
#pragma unroll
    for (int nt = 0; nt < 2; ++nt) {
      v8f z = {0.f, 0.f, 0.f, 0.f, 0.f, 0.f, 0.f, 0.f};
      acc[mt][nt] = z;
    }

#if USE_TDM
  // Wave 0 issues two TDM descriptors per k-step (A tile, B tile).
  const tdm_g1 g1 = make_g1_2d(/*tensor_d0=*/DIM, /*tensor_d1=*/S_LEN,
                               /*tile_d0=*/32, /*tile_d1=*/128,
                               /*stride0=*/DIM);
  const tdm_g2 zg4 = {0, 0, 0, 0};
  const tdm_g1 zg8 = {0, 0, 0, 0, 0, 0, 0, 0};
  auto issue_tile = [&](int kk, int bufi) {
    const unsigned long long ga =
        (unsigned long long)(uintptr_t)Abf + ((size_t)m0 * DIM + kk * 32) * 2ull;
    const unsigned long long gb =
        (unsigned long long)(uintptr_t)Wbf + ((size_t)n0 * DIM + kk * 32) * 2ull;
    const unsigned la = (unsigned)(uintptr_t)&sA[bufi][0];
    const unsigned lb = (unsigned)(uintptr_t)&sB[bufi][0];
    __builtin_amdgcn_tensor_load_to_lds(make_g0(la, ga), g1, zg4, zg4, zg8, 0);
    __builtin_amdgcn_tensor_load_to_lds(make_g0(lb, gb), g1, zg4, zg4, zg8, 0);
  };

  if (tid < 32) {
    issue_tile(0, 0);
    __builtin_amdgcn_s_wait_tensorcnt(0);
  }
  __syncthreads();
#else
  // Fallback: plain vector loads of the pre-converted bf16 tiles.
  auto load_tile = [&](int kk, int bufi) {
    const int k0e = kk * 32;
#pragma unroll
    for (int j = tid; j < 512; j += 256) {   // 512 x 16B chunks per tile
      int row = j >> 2;
      int c8  = (j & 3) * 8;
      *(uint4*)&sA[bufi][row * 32 + c8] =
          *(const uint4*)(Abf + (size_t)(m0 + row) * DIM + k0e + c8);
      *(uint4*)&sB[bufi][row * 32 + c8] =
          *(const uint4*)(Wbf + (size_t)(n0 + row) * DIM + k0e + c8);
    }
  };
  load_tile(0, 0);
  __syncthreads();
#endif

  int buf = 0;
  for (int kk = 0; kk < DIM / 32; ++kk) {
#if USE_TDM
    if (kk + 1 < DIM / 32 && tid < 32) issue_tile(kk + 1, buf ^ 1);
#else
    if (kk + 1 < DIM / 32) load_tile(kk + 1, buf ^ 1);
#endif

    v16bf afrag[4], bfrag[2];
#pragma unroll
    for (int mt = 0; mt < 4; ++mt)
      afrag[mt] = load_frag(&sA[buf][0], wm * 64 + mt * 16 + (lane & 15), lane);
#pragma unroll
    for (int nt = 0; nt < 2; ++nt)
      bfrag[nt] = load_frag(&sB[buf][0], wn * 32 + nt * 16 + (lane & 15), lane);

#pragma unroll
    for (int mt = 0; mt < 4; ++mt)
#pragma unroll
      for (int nt = 0; nt < 2; ++nt)
        acc[mt][nt] = __builtin_amdgcn_wmma_f32_16x16x32_bf16(
            false, afrag[mt], false, bfrag[nt],
            (short)0, acc[mt][nt], false, false);

#if USE_TDM
    if (tid < 32) __builtin_amdgcn_s_wait_tensorcnt(0);
#endif
    __syncthreads();
    buf ^= 1;
  }

  // Epilogue: D layout -> lanes 0-15 cols, VGPR r -> row r (lanes>=16: +8)
#pragma unroll
  for (int nt = 0; nt < 2; ++nt) {
    const int col  = n0 + wn * 32 + nt * 16 + (lane & 15);
    const float bias = b_ih[col] + b_hh[col];
#pragma unroll
    for (int mt = 0; mt < 4; ++mt) {
      const int rbase = m0 + wm * 64 + mt * 16 + ((lane >> 4) << 3);
#pragma unroll
      for (int r = 0; r < 8; ++r)
        xg[(size_t)(rbase + r) * G4 + col] = acc[mt][nt][r] + bias;
    }
  }
}

// ---------------------------------------------------------------------------
// Grid-wide barrier for the persistent scan kernel (16 co-resident WGs).
// Cluster barrier is an architectural NOP when not cluster-dispatched, but
// uses the CDNA5 path (s_barrier_signal/wait -3) if the launch is clustered.
// ---------------------------------------------------------------------------
static __device__ __forceinline__ void grid_sync(int* cnt, int* gen, int& genl) {
  __threadfence();
  __syncthreads();
  if (threadIdx.x == 0) {
    int target = ++genl;
    if (atomicAdd(cnt, 1) == NWG - 1) {
      atomicExch(cnt, 0);
      __threadfence();
      atomicExch(gen, target);
    } else {
      while (atomicAdd(gen, 0) < target) __builtin_amdgcn_s_sleep(1);
    }
  } else {
    ++genl;
  }
  __builtin_amdgcn_s_cluster_barrier();
  __syncthreads();
}

// ---------------------------------------------------------------------------
// Kernel 2: sequential LSTM scan. 16 WGs x 1024 threads.
// WG w owns hidden units [w*64, w*64+64); wave v owns units u0=w*64+v*2, +1
// and their 4 gate rows each (8 GEMV rows / wave / step). h broadcast via
// global double buffer + grid barrier; per-unit c state in lane registers.
// ---------------------------------------------------------------------------
__global__ void __launch_bounds__(1024)
lstm_scan(const float* __restrict__ xg,    // [16384,4096]
          const float* __restrict__ W_hh,  // [4096,1024]
          const float* __restrict__ W_fc,  // [1024]
          float* __restrict__ hglob,       // [2,1024]
          float* __restrict__ partial,     // [16384,16]
          int* __restrict__ syncs)
{
  __shared__ float sh[HID];
  __shared__ float alpha_arr[64];

  const int tid  = threadIdx.x;
  const int w    = blockIdx.x;           // 0..15
  const int wave = tid >> 5;             // 0..31
  const int lane = tid & 31;
  const int u0   = w * 64 + wave * 2;    // first of this wave's 2 units

  int*  cnt = &syncs[0];
  int*  gen = &syncs[1];
  int   genl = 0;
  float creg = 0.f;                      // c state (lanes 0,1 meaningful)

  sh[tid] = 0.f;                         // h_{-1} = 0
  __syncthreads();

  for (int t = 0; t < S_LEN; ++t) {
    if (lane == 0 && t + 1 < S_LEN)
      __builtin_prefetch(xg + (size_t)(t + 1) * G4 + u0, 0, 1);

    // preload this lane's fixed k-slice of h (32 floats)
    float4 hreg[8];
#pragma unroll
    for (int kb = 0; kb < 8; ++kb)
      hreg[kb] = *(const float4*)&sh[kb * 128 + lane * 4];

    // 8 row-dots: unit j in {0,1} x gate gi in {0..3}
    float s[2][4];
#pragma unroll
    for (int j = 0; j < 2; ++j) {
#pragma unroll
      for (int gi = 0; gi < 4; ++gi) {
        const float* wrow = W_hh + (size_t)(u0 + j + gi * HID) * HID;
        float acc = 0.f;
#pragma unroll
        for (int kb = 0; kb < 8; ++kb) {
          float4 wv = *(const float4*)&wrow[kb * 128 + lane * 4];
          acc += wv.x * hreg[kb].x + wv.y * hreg[kb].y +
                 wv.z * hreg[kb].z + wv.w * hreg[kb].w;
        }
#pragma unroll
        for (int off = 16; off > 0; off >>= 1)
          acc += __shfl_xor(acc, off, 32);
        s[j][gi] = acc;                  // full sum on all lanes
      }
    }

    if (lane < 2) {                      // lane j handles unit u0+j
      const int u = u0 + lane;
      const float* xr = xg + (size_t)t * G4;
      float sg0 = (lane == 0) ? s[0][0] : s[1][0];
      float sg1 = (lane == 0) ? s[0][1] : s[1][1];
      float sg2 = (lane == 0) ? s[0][2] : s[1][2];
      float sg3 = (lane == 0) ? s[0][3] : s[1][3];
      float gi_ = xr[u           ] + sg0;
      float gf_ = xr[u + 1 * HID ] + sg1;
      float gg_ = xr[u + 2 * HID ] + sg2;
      float go_ = xr[u + 3 * HID ] + sg3;
      float ii = 1.f / (1.f + __expf(-gi_));
      float ff = 1.f / (1.f + __expf(-gf_));
      float oo = 1.f / (1.f + __expf(-go_));
      creg = ff * creg + ii * tanhf(gg_);
      float hnew = oo * tanhf(creg);
      hglob[(t & 1) * HID + u] = hnew;
      alpha_arr[wave * 2 + lane] = hnew * W_fc[u];
    }

    grid_sync(cnt, gen, genl);           // h_t fully published device-wide

    if (tid == 0) {                      // WG-partial of h_t . W_fc
      float a = 0.f;
#pragma unroll
      for (int i = 0; i < 64; ++i) a += alpha_arr[i];
      partial[(size_t)t * NWG + w] = a;
    }
    sh[tid] = hglob[(t & 1) * HID + tid]; // gather full h_t
    __syncthreads();
  }
}

// ---------------------------------------------------------------------------
// Kernel 3: alpha[t] = sigmoid(sum_w partial[t,w] + b_fc)
// ---------------------------------------------------------------------------
__global__ void alpha_fin(const float* __restrict__ partial,
                          const float* __restrict__ b_fc,
                          float* __restrict__ alpha)
{
  int t = blockIdx.x * blockDim.x + threadIdx.x;
  if (t < S_LEN) {
    float s = b_fc[0];
#pragma unroll
    for (int i = 0; i < NWG; ++i) s += partial[(size_t)t * NWG + i];
    alpha[t] = 1.f / (1.f + __expf(-s));
  }
}

// ---------------------------------------------------------------------------
// Kernel 4: the a_g / y recurrences (affine scans) + loss. One WG, 1024 thr.
//   x_i = m_i * x_{i-1} + b_i ; compose (m2,b2)o(m1,b1) = (m1*m2, m2*b1+b2)
// ---------------------------------------------------------------------------
__global__ void __launch_bounds__(1024)
rec_kernel(const float* __restrict__ alpha,
           const float* __restrict__ uttr_pred,
           const float* __restrict__ timing_label,
           float* __restrict__ out)     // [1 + 3*REC_N]
{
  __shared__ float sM[1024], sB[1024];
  __shared__ int firstIdx;
  const int tid = threadIdx.x;
  const int CH  = 16;                    // 1024*16 >= 16383
  float* y_out = out + 1;
  float* a_out = out + 1 + REC_N;
  float* u_out = out + 1 + 2 * REC_N;

  for (int i = tid; i < REC_N; i += 1024)
    u_out[i] = 1.f - uttr_pred[i + 1];
  __threadfence_block();
  __syncthreads();

  for (int phase = 0; phase < 2; ++phase) {
    const int i0 = tid * CH;
    float M = 1.f, B = 0.f;
    for (int k = 0; k < CH; ++k) {
      int i = i0 + k;
      float m = 1.f, b = 0.f;
      if (i < REC_N) {
        float u = u_out[i];
        if (phase == 0) { m = u; b = (1.f - u) * alpha[i + 1]; }
        else            { float ag = a_out[i]; m = 1.f - ag; b = ag * u; }
      }
      M = m * M;
      B = m * B + b;
    }
    sM[tid] = M; sB[tid] = B;
    __syncthreads();
    for (int off = 1; off < 1024; off <<= 1) {
      float m2, b2;
      if (tid >= off) { m2 = sM[tid] * sM[tid - off];
                        b2 = sM[tid] * sB[tid - off] + sB[tid]; }
      else            { m2 = sM[tid]; b2 = sB[tid]; }
      __syncthreads();
      sM[tid] = m2; sB[tid] = b2;
      __syncthreads();
    }
    float x = (tid == 0) ? 0.f : sB[tid - 1];
    for (int k = 0; k < CH; ++k) {
      int i = i0 + k;
      if (i < REC_N) {
        float u = u_out[i];
        float m, b;
        if (phase == 0) { m = u; b = (1.f - u) * alpha[i + 1]; }
        else            { float ag = a_out[i]; m = 1.f - ag; b = ag * u; }
        x = m * x + b;
        if (phase == 0) a_out[i] = x; else y_out[i] = x;
      }
    }
    __threadfence_block();
    __syncthreads();
  }

  if (tid == 0) firstIdx = REC_N;
  __syncthreads();
  for (int i = tid; i < REC_N; i += 1024)
    if (timing_label[i] > 0.8f) atomicMin(&firstIdx, i);
  __syncthreads();
  if (tid == 0) {
    float loss;
    float yend = y_out[REC_N - 1];
    if (firstIdx < REC_N) {
      float uat = u_out[firstIdx];
      float yat = y_out[firstIdx];
      float d = yat - 0.8f;
      loss = (uat < 0.5f) ? 0.f : d * d;
    } else {
      float d = yend - 0.4f;
      loss = (yend >= 0.8f) ? d * d : 0.f;
    }
    out[0] = loss;
  }
}

// ---------------------------------------------------------------------------
extern "C" void kernel_launch(void* const* d_in, const int* in_sizes, int n_in,
                              void* d_out, int out_size, void* d_ws, size_t ws_size,
                              hipStream_t stream) {
  (void)in_sizes; (void)n_in; (void)out_size; (void)ws_size;
  const float* inputs       = (const float*)d_in[0];
  const float* uttr_pred    = (const float*)d_in[1];
  const float* timing_label = (const float*)d_in[2];
  // d_in[3] uttr_label: unused by the returned quantities
  const float* W_ih         = (const float*)d_in[4];
  const float* W_hh         = (const float*)d_in[5];
  const float* b_ih         = (const float*)d_in[6];
  const float* b_hh         = (const float*)d_in[7];
  const float* W_fc         = (const float*)d_in[8];
  const float* b_fc         = (const float*)d_in[9];
  float* out = (float*)d_out;

  float* ws      = (float*)d_ws;
  float* xg      = ws;                                    // 16384*4096 f
  float* hglob   = xg + (size_t)S_LEN * G4;               // 2*1024 f
  float* partial = hglob + 2 * HID;                       // 16384*16 f
  float* alpha   = partial + (size_t)S_LEN * NWG;         // 16384 f
  int*   syncs   = (int*)(alpha + S_LEN);                 // 4 ints
  unsigned short* Abf = (unsigned short*)(syncs + 4);     // 16384*1024 bf16
  unsigned short* Wbf = Abf + (size_t)S_LEN * DIM;        // 4096*1024 bf16

  init_kernel<<<1, 256, 0, stream>>>(syncs, hglob);
  cvt_bf16<<<(S_LEN * DIM) / (256 * 8), 256, 0, stream>>>(inputs, Abf, S_LEN * DIM);
  cvt_bf16<<<(G4 * DIM) / (256 * 8), 256, 0, stream>>>(W_ih, Wbf, G4 * DIM);
  gemm_xg<<<dim3(G4 / 128, S_LEN / 128), 256, 0, stream>>>(
      Abf, Wbf, b_ih, b_hh, xg);
  lstm_scan<<<NWG, 1024, 0, stream>>>(xg, W_hh, W_fc, hglob, partial, syncs);
  alpha_fin<<<S_LEN / 256, 256, 0, stream>>>(partial, b_fc, alpha);
  rec_kernel<<<1, 1024, 0, stream>>>(alpha, uttr_pred, timing_label, out);
}
